// SGGA_49589692400289
// MI455X (gfx1250) — matrix operand order repaired and checked by reading, not compiled, not run on hardware
//
#include <hip/hip_runtime.h>

// ---------------------------------------------------------------------------
// CDNA5 (gfx1250) transformer block: 3x MHA + FFN + LayerNorms.
// GEMMs / attention contractions use v_wmma_f32_16x16x32_bf16 (wave32).
// LDS strides padded to 16B alignment so fragment loads are ds_load_b128.
// Global->LDS tile movement is register-staged + LDS double-buffered so all
// global_load_b128 of a tile are in flight together and hide behind WMMAs.
// ---------------------------------------------------------------------------

typedef __attribute__((ext_vector_type(16))) __bf16 v16bf;
typedef __attribute__((ext_vector_type(8)))  float  v8f;

#define WMMA_BF16(a, b, c) \
    __builtin_amdgcn_wmma_f32_16x16x32_bf16(false, (a), false, (b), (short)0, (c), false, false)

// --- Fragment loaders (CDNA5 WMMA VGPR layouts, CDNA5 ISA) -------
// A 16x32 bf16 (MxK): lane L holds row M=L&15.
//   lanes 0-15:  VGPR0..3 -> K=0..7,  VGPR4..7 -> K=16..23
//   lanes 16-31: VGPR0..3 -> K=8..15, VGPR4..7 -> K=24..31
// Per lane this is two contiguous 16B chunks at +0 and +32 bytes.
__device__ __forceinline__ v16bf load_frag_a(const __bf16* base, int stride, int kofs, int lane)
{
    union { v16bf v; uint4 q[2]; } f;
    const __bf16* p = base + (size_t)(lane & 15) * stride + kofs + ((lane >> 4) << 3);
    f.q[0] = *(const uint4*)(p);
    f.q[1] = *(const uint4*)(p + 16);
    return f.v;
}

// B 32x16 bf16 (KxN): lane L holds column N=L&15.
//   lanes 0-15:  VGPR0..3 -> K=0..7,  VGPR4..7 -> K=16..23
//   lanes 16-31: VGPR0..3 -> K=8..15, VGPR4..7 -> K=24..31
// Per lane this is two contiguous 16B chunks at +0 and +32 bytes.
__device__ __forceinline__ v16bf load_frag_b(const __bf16* base, int stride, int kofs, int lane)
{
    union { v16bf v; uint4 q[2]; } f;
    const __bf16* p = base + (size_t)(lane & 15) * stride + kofs + ((lane >> 4) << 3);
    f.q[0] = *(const uint4*)(p);
    f.q[1] = *(const uint4*)(p + 16);
    return f.v;
}

// C/D 16x16 f32: element i -> row M = i + 8*(lane>=16), col N = lane&15.

// ---------------------------------------------------------------------------
// GEMM: C[M,N] = A[M,K](bf16) @ W[K,N](bf16, pre-transposed as Wt[N][K]) + bias.
// Block tile 128x128, 8 waves (4 M x 2 N), each wave 32x64 (2x4 WMMA frags)
// -> 8 v_wmma per 32-wide K step. LDS ping-pong, one barrier per step.
// ---------------------------------------------------------------------------
__global__ __launch_bounds__(256)
void gemm_bf16_kernel(const __bf16* __restrict__ A, const __bf16* __restrict__ Wt,
                      const float* __restrict__ bias,
                      float* __restrict__ Cf, __bf16* __restrict__ Cb,
                      int M, int N, int K, int relu)
{
    (void)M;
    __shared__ __align__(16) __bf16 As[2][128 * 40];   // [m][k], 80B row stride
    __shared__ __align__(16) __bf16 Bs[2][128 * 40];   // [n][k], 80B row stride
    const int tid = threadIdx.x, lane = tid & 31, wid = tid >> 5;
    const int wm = (wid & 3) * 32, wn = (wid >> 2) * 64;
    const int tileM = blockIdx.x * 128, tileN = blockIdx.y * 128;

    // per-thread tile-copy coordinates (128x32 bf16 = 512 uint4, 2 per thread)
    const int rT = tid >> 2, cT = (tid & 3) * 8;

    uint4 ra0, ra1, rb0, rb1;
    auto fetch = [&](int k0) {
        ra0 = *(const uint4*)(A  + (size_t)(tileM + rT) * K + k0 + cT);
        ra1 = *(const uint4*)(A  + (size_t)(tileM + rT + 64) * K + k0 + cT);
        rb0 = *(const uint4*)(Wt + (size_t)(tileN + rT) * K + k0 + cT);
        rb1 = *(const uint4*)(Wt + (size_t)(tileN + rT + 64) * K + k0 + cT);
    };
    auto stash = [&](int buf) {
        *(uint4*)(As[buf] + rT * 40 + cT)        = ra0;
        *(uint4*)(As[buf] + (rT + 64) * 40 + cT) = ra1;
        *(uint4*)(Bs[buf] + rT * 40 + cT)        = rb0;
        *(uint4*)(Bs[buf] + (rT + 64) * 40 + cT) = rb1;
    };

    v8f zero = {};
    v8f c[2][4];
#pragma unroll
    for (int mi = 0; mi < 2; ++mi)
#pragma unroll
        for (int ni = 0; ni < 4; ++ni) c[mi][ni] = zero;

    fetch(0);
    stash(0);
    __syncthreads();
    int cur = 0;

    for (int k0 = 0; k0 < K; k0 += 32) {
        const bool nxt = (k0 + 32) < K;
        if (nxt) fetch(k0 + 32);                    // next tile: loads in flight
        if (k0 + 64 < K) {                          // k+2 tile: global_prefetch_b8
            __builtin_prefetch(A  + (size_t)(tileM + (tid >> 1)) * K + k0 + 64, 0, 1);
            __builtin_prefetch(Wt + (size_t)(tileN + (tid >> 1)) * K + k0 + 64, 0, 1);
        }

        const __bf16* Ab = As[cur];
        const __bf16* Bb = Bs[cur];
        v16bf a0 = load_frag_a(Ab + (size_t)wm * 40, 40, 0, lane);
        v16bf a1 = load_frag_a(Ab + (size_t)(wm + 16) * 40, 40, 0, lane);
        v16bf b0 = load_frag_b(Bb + (size_t)(wn)*40, 40, 0, lane);
        v16bf b1 = load_frag_b(Bb + (size_t)(wn + 16) * 40, 40, 0, lane);
        v16bf b2 = load_frag_b(Bb + (size_t)(wn + 32) * 40, 40, 0, lane);
        v16bf b3 = load_frag_b(Bb + (size_t)(wn + 48) * 40, 40, 0, lane);

        c[0][0] = WMMA_BF16(a0, b0, c[0][0]);
        c[0][1] = WMMA_BF16(a0, b1, c[0][1]);
        c[0][2] = WMMA_BF16(a0, b2, c[0][2]);
        c[0][3] = WMMA_BF16(a0, b3, c[0][3]);
        c[1][0] = WMMA_BF16(a1, b0, c[1][0]);
        c[1][1] = WMMA_BF16(a1, b1, c[1][1]);
        c[1][2] = WMMA_BF16(a1, b2, c[1][2]);
        c[1][3] = WMMA_BF16(a1, b3, c[1][3]);

        if (nxt) stash(cur ^ 1);                    // fill other buffer
        __syncthreads();
        cur ^= 1;
    }

    // Epilogue: bias/relu folded into accumulators, then uniform store branches.
    const int hl = lane >> 4, nlo = lane & 15;
#pragma unroll
    for (int ni = 0; ni < 4; ++ni) {
        float bval = bias[tileN + wn + ni * 16 + nlo];
#pragma unroll
        for (int mi = 0; mi < 2; ++mi)
#pragma unroll
            for (int i = 0; i < 8; ++i) c[mi][ni][i] += bval;
    }
    if (relu) {
#pragma unroll
        for (int mi = 0; mi < 2; ++mi)
#pragma unroll
            for (int ni = 0; ni < 4; ++ni)
#pragma unroll
                for (int i = 0; i < 8; ++i) c[mi][ni][i] = fmaxf(c[mi][ni][i], 0.0f);
    }
    if (Cf) {
#pragma unroll
        for (int mi = 0; mi < 2; ++mi)
#pragma unroll
            for (int ni = 0; ni < 4; ++ni)
#pragma unroll
                for (int i = 0; i < 8; ++i) {
                    int row = tileM + wm + mi * 16 + i + 8 * hl;
                    int col = tileN + wn + ni * 16 + nlo;
                    Cf[(size_t)row * N + col] = c[mi][ni][i];
                }
    }
    if (Cb) {
#pragma unroll
        for (int mi = 0; mi < 2; ++mi)
#pragma unroll
            for (int ni = 0; ni < 4; ++ni)
#pragma unroll
                for (int i = 0; i < 8; ++i) {
                    int row = tileM + wm + mi * 16 + i + 8 * hl;
                    int col = tileN + wn + ni * 16 + nlo;
                    Cb[(size_t)row * N + col] = (__bf16)c[mi][ni][i];
                }
    }
}

// ---------------------------------------------------------------------------
// Flash attention: per (b, h), each wave owns 32 query rows, streams K/V in
// 32-key tiles with online softmax. Q/K/V: bf16 [b, l, H*64] (head h = cols
// h*64..h*64+63). mask[b, lk]==0 -> score = -1e9 (matches reference).
// K/V tiles are register-staged + LDS double-buffered.
// ---------------------------------------------------------------------------
__global__ __launch_bounds__(256)
void attn_kernel(const __bf16* __restrict__ Q, const __bf16* __restrict__ Kg,
                 const __bf16* __restrict__ Vg, const int* __restrict__ mask,
                 __bf16* __restrict__ O, int Lq, int Lk)
{
    __shared__ __align__(16) __bf16 Ks[2][32 * 72];   // [key][dh]  B-frags for S
    __shared__ __align__(16) __bf16 Vt[2][64 * 40];   // [dh][key]  B-frags for P@V
    __shared__ __align__(16) __bf16 Ps[8 * 32 * 40];  // per-wave P scratch [q][key]
    const int b = blockIdx.z, h = blockIdx.y;
    const int tid = threadIdx.x, lane = tid & 31, w = tid >> 5;
    const int hl = lane >> 4, nlo = lane & 15;
    const int qbase = blockIdx.x * 256 + w * 32;
    const float scale = 0.125f;                       // 1/sqrt(64)

    // K-tile copy coords: 32x64 bf16 = 256 uint4, 1 per thread
    const int rK = tid >> 3, cK = (tid & 7) * 8;

    uint4 kreg;
    unsigned int vreg[4];
    auto fetchKV = [&](int kt) {
        kreg = *(const uint4*)(Kg + ((size_t)(b * Lk + kt + rK)) * 512 + h * 64 + cK);
#pragma unroll
        for (int t = 0; t < 4; ++t) {
            int idx = tid + t * 256;
            int rr = idx >> 5, c2 = idx & 31;                 // key, dh-pair
            vreg[t] = *(const unsigned int*)(
                Vg + ((size_t)(b * Lk + kt + rr)) * 512 + h * 64 + c2 * 2);
        }
    };
    auto stashKV = [&](int buf) {
        *(uint4*)(Ks[buf] + rK * 72 + cK) = kreg;
#pragma unroll
        for (int t = 0; t < 4; ++t) {
            int idx = tid + t * 256;
            int rr = idx >> 5, c2 = idx & 31;
            union { unsigned int u; __bf16 h2[2]; } uu;
            uu.u = vreg[t];
            Vt[buf][(c2 * 2) * 40 + rr]     = uu.h2[0];
            Vt[buf][(c2 * 2 + 1) * 40 + rr] = uu.h2[1];
        }
    };

    // Q fragments (register-resident for the whole kernel)
    v16bf qa[2][2];
#pragma unroll
    for (int mi = 0; mi < 2; ++mi)
#pragma unroll
        for (int kc = 0; kc < 2; ++kc)
            qa[mi][kc] = load_frag_a(
                Q + ((size_t)(b * Lq + qbase + mi * 16)) * 512 + h * 64, 512, kc * 32, lane);

    v8f zero = {};
    v8f o[2][4];
    float mrow[2][8], lrow[2][8];
#pragma unroll
    for (int mi = 0; mi < 2; ++mi) {
#pragma unroll
        for (int nj = 0; nj < 4; ++nj) o[mi][nj] = zero;
#pragma unroll
        for (int i = 0; i < 8; ++i) { mrow[mi][i] = -1e30f; lrow[mi][i] = 0.0f; }
    }
    __bf16* Pw = Ps + (size_t)w * 32 * 40;

    fetchKV(0);
    stashKV(0);
    __syncthreads();
    int cur = 0;

    for (int kt = 0; kt < Lk; kt += 32) {
        const bool nxt = (kt + 32) < Lk;
        if (nxt) fetchKV(kt + 32);                  // next tile loads in flight

        const __bf16* Kb = Ks[cur];
        const __bf16* Vb = Vt[cur];

        // S = Q @ K^T  (M=q, N=key, K=dh)
        v8f s[2][2];
        s[0][0] = zero; s[0][1] = zero; s[1][0] = zero; s[1][1] = zero;
#pragma unroll
        for (int mi = 0; mi < 2; ++mi)
#pragma unroll
            for (int ni = 0; ni < 2; ++ni)
#pragma unroll
                for (int kc = 0; kc < 2; ++kc)
                    s[mi][ni] = WMMA_BF16(qa[mi][kc],
                        load_frag_b(Kb + (size_t)ni * 16 * 72, 72, kc * 32, lane), s[mi][ni]);

        // scale + mask
        int mk0 = mask[b * Lk + kt + nlo];
        int mk1 = mask[b * Lk + kt + 16 + nlo];
#pragma unroll
        for (int mi = 0; mi < 2; ++mi)
#pragma unroll
            for (int i = 0; i < 8; ++i) {
                s[mi][0][i] = mk0 ? s[mi][0][i] * scale : -1e9f;
                s[mi][1][i] = mk1 ? s[mi][1][i] * scale : -1e9f;
            }

        // online softmax per 16-row group; write P (bf16) to LDS scratch
#pragma unroll
        for (int mi = 0; mi < 2; ++mi) {
            float mx[8], rs[8];
#pragma unroll
            for (int i = 0; i < 8; ++i) mx[i] = fmaxf(s[mi][0][i], s[mi][1][i]);
#pragma unroll
            for (int off = 1; off < 16; off <<= 1)
#pragma unroll
                for (int i = 0; i < 8; ++i) mx[i] = fmaxf(mx[i], __shfl_xor(mx[i], off, 32));
#pragma unroll
            for (int i = 0; i < 8; ++i) {
                float mn = fmaxf(mrow[mi][i], mx[i]);
                float alpha = __expf(mrow[mi][i] - mn);
                mrow[mi][i] = mn;
                lrow[mi][i] *= alpha;
#pragma unroll
                for (int nj = 0; nj < 4; ++nj) o[mi][nj][i] *= alpha;
                float p0 = __expf(s[mi][0][i] - mn);
                float p1 = __expf(s[mi][1][i] - mn);
                rs[i] = p0 + p1;
                int rloc = mi * 16 + i + 8 * hl;
                Pw[rloc * 40 + nlo]      = (__bf16)p0;
                Pw[rloc * 40 + 16 + nlo] = (__bf16)p1;
            }
#pragma unroll
            for (int off = 1; off < 16; off <<= 1)
#pragma unroll
                for (int i = 0; i < 8; ++i) rs[i] += __shfl_xor(rs[i], off, 32);
#pragma unroll
            for (int i = 0; i < 8; ++i) lrow[mi][i] += rs[i];
        }

        // O += P @ V  (M=q, K=key, N=dh)
#pragma unroll
        for (int mi = 0; mi < 2; ++mi) {
            v16bf pa = load_frag_a(Pw + (size_t)mi * 16 * 40, 40, 0, lane);
#pragma unroll
            for (int nj = 0; nj < 4; ++nj)
                o[mi][nj] = WMMA_BF16(pa,
                    load_frag_b(Vb + (size_t)nj * 16 * 40, 40, 0, lane), o[mi][nj]);
        }

        if (nxt) stashKV(cur ^ 1);
        __syncthreads();
        cur ^= 1;
    }

    // normalize by softmax denominator and store bf16 output [b, q, h*64+dh]
#pragma unroll
    for (int mi = 0; mi < 2; ++mi)
#pragma unroll
        for (int i = 0; i < 8; ++i) {
            float inv = 1.0f / fmaxf(lrow[mi][i], 1e-20f);
            int row = qbase + mi * 16 + i + 8 * hl;
            size_t base = ((size_t)(b * Lq + row)) * 512 + h * 64;
#pragma unroll
            for (int nj = 0; nj < 4; ++nj)
                O[base + nj * 16 + nlo] = (__bf16)(o[mi][nj][i] * inv);
        }
}

// ---------------------------------------------------------------------------
// LayerNorm over D=512 (one block per row) with fused residual add.
// ---------------------------------------------------------------------------
__global__ __launch_bounds__(256)
void ln_kernel(const float* __restrict__ X, const float* __restrict__ R,
               const float* __restrict__ g, const float* __restrict__ be,
               float* __restrict__ Yf, __bf16* __restrict__ Yb)
{
    __shared__ float sh1[256];
    __shared__ float sh2[256];
    int row = blockIdx.x, tid = threadIdx.x;
    size_t base = (size_t)row * 512;
    float v0 = X[base + tid]       + (R ? R[base + tid]       : 0.0f);
    float v1 = X[base + 256 + tid] + (R ? R[base + 256 + tid] : 0.0f);
    sh1[tid] = v0 + v1;
    sh2[tid] = v0 * v0 + v1 * v1;
    __syncthreads();
    for (int s = 128; s > 0; s >>= 1) {
        if (tid < s) { sh1[tid] += sh1[tid + s]; sh2[tid] += sh2[tid + s]; }
        __syncthreads();
    }
    float mean = sh1[0] * (1.0f / 512.0f);
    float var  = sh2[0] * (1.0f / 512.0f) - mean * mean;
    float rstd = rsqrtf(var + 1e-6f);
    float y0 = (v0 - mean) * rstd * g[tid]       + be[tid];
    float y1 = (v1 - mean) * rstd * g[256 + tid] + be[256 + tid];
    Yf[base + tid] = y0;
    Yf[base + 256 + tid] = y1;
    if (Yb) { Yb[base + tid] = (__bf16)y0; Yb[base + 256 + tid] = (__bf16)y1; }
}

__global__ __launch_bounds__(256)
void add_kernel(const float* __restrict__ A, const float* __restrict__ B,
                float* __restrict__ Cf, __bf16* __restrict__ Cb, size_t n)
{
    size_t i0 = (size_t)blockIdx.x * 1024 + threadIdx.x;
#pragma unroll
    for (int t = 0; t < 4; ++t) {
        size_t j = i0 + (size_t)t * 256;
        if (j < n) {
            float v = A[j] + B[j];
            Cf[j] = v;
            Cb[j] = (__bf16)v;
        }
    }
}

__global__ __launch_bounds__(256)
void cvt_kernel(const float* __restrict__ X, __bf16* __restrict__ Y, size_t n)
{
    size_t i0 = (size_t)blockIdx.x * 1024 + threadIdx.x;
#pragma unroll
    for (int t = 0; t < 4; ++t) {
        size_t j = i0 + (size_t)t * 256;
        if (j < n) Y[j] = (__bf16)X[j];
    }
}

// f32 W[K][N] -> bf16 Wt[N][K] (weights converted+transposed once per launch)
__global__ __launch_bounds__(256)
void cvtT_kernel(const float* __restrict__ X, __bf16* __restrict__ Y, int Kd, int Nd)
{
    size_t n = (size_t)Kd * Nd;
    size_t i0 = (size_t)blockIdx.x * 1024 + threadIdx.x;
#pragma unroll
    for (int t = 0; t < 4; ++t) {
        size_t j = i0 + (size_t)t * 256;
        if (j < n) {
            int k = (int)(j / Nd), nn = (int)(j % Nd);
            Y[(size_t)nn * Kd + k] = (__bf16)X[j];
        }
    }
}

// ---------------------------------------------------------------------------
// Host orchestration
// ---------------------------------------------------------------------------
extern "C" void kernel_launch(void* const* d_in, const int* in_sizes, int n_in,
                              void* d_out, int out_size, void* d_ws, size_t ws_size,
                              hipStream_t stream)
{
    (void)in_sizes; (void)n_in; (void)out_size; (void)ws_size;
    const int Dm = 512, FFd = 2048, Bn = 8, L = 1024;
    const int M = Bn * L;                       // 8192 rows
    const size_t MD  = (size_t)M * Dm;          // activations elems
    const size_t DD  = (size_t)Dm * Dm;         // 512x512 weight elems
    const size_t DF  = (size_t)Dm * FFd;        // 512x2048

    // input indices per setup_inputs() dict insertion order
    const float* x  = (const float*)d_in[0];
    const int* x_mask  = (const int*)d_in[3];
    const int* y1_mask = (const int*)d_in[4];
    const int* y2_mask = (const int*)d_in[5];
    const int mha_base[3] = {6, 14, 22};        // sa, gaa, gab: wq,bq,wk,bk,wv,bv,wm,bm
    const float* ffn_w1 = (const float*)d_in[30];
    const float* ffn_b1 = (const float*)d_in[31];
    const float* ffn_w2 = (const float*)d_in[32];
    const float* ffn_b2 = (const float*)d_in[33];
    const float* ln_g[4] = {(const float*)d_in[34], (const float*)d_in[36],
                            (const float*)d_in[38], (const float*)d_in[40]};
    const float* ln_b[4] = {(const float*)d_in[35], (const float*)d_in[37],
                            (const float*)d_in[39], (const float*)d_in[41]};

    // workspace bump allocator
    char* wp = (char*)d_ws;
    auto alloc = [&](size_t bytes) { char* r = wp; wp += (bytes + 255) & ~(size_t)255; return r; };

    __bf16* xb   = (__bf16*)alloc(MD * 2);
    __bf16* y1b  = (__bf16*)alloc(MD * 2);
    __bf16* y2b  = (__bf16*)alloc(MD * 2);
    __bf16* qb   = (__bf16*)alloc(MD * 2);
    __bf16* kb   = (__bf16*)alloc(MD * 2);
    __bf16* vb   = (__bf16*)alloc(MD * 2);
    __bf16* attnb= (__bf16*)alloc(MD * 2);
    float*  projf= (float*)alloc(MD * 4);
    float*  x1f  = (float*)alloc(MD * 4);
    __bf16* x1b  = (__bf16*)alloc(MD * 2);
    float*  yx1f = (float*)alloc(MD * 4);
    float*  yx2f = (float*)alloc(MD * 4);
    float*  xuf  = (float*)alloc(MD * 4);
    __bf16* xub  = (__bf16*)alloc(MD * 2);
    __bf16* h1b  = (__bf16*)alloc((size_t)M * FFd * 2);
    float*  h2f  = projf;                       // reuse (proj GEMMs are done by then)
    __bf16* w1t  = (__bf16*)alloc(DF * 2);      // [FF][D]
    __bf16* w2t  = (__bf16*)alloc(DF * 2);      // [D][FF]
    struct MhaW { __bf16 *wq, *wk, *wv, *wm; } mw[3];
    for (int m3 = 0; m3 < 3; ++m3) {
        mw[m3].wq = (__bf16*)alloc(DD * 2);
        mw[m3].wk = (__bf16*)alloc(DD * 2);
        mw[m3].wv = (__bf16*)alloc(DD * 2);
        mw[m3].wm = (__bf16*)alloc(DD * 2);
    }

    auto cvt = [&](const float* src, __bf16* dst, size_t n) {
        int blocks = (int)((n + 1023) / 1024);
        cvt_kernel<<<blocks, 256, 0, stream>>>(src, dst, n);
    };
    auto cvtT = [&](const float* src, __bf16* dst, int Kd, int Nd) {
        size_t n = (size_t)Kd * Nd;
        int blocks = (int)((n + 1023) / 1024);
        cvtT_kernel<<<blocks, 256, 0, stream>>>(src, dst, Kd, Nd);
    };
    auto gemm = [&](const __bf16* A, const __bf16* Wt, const float* bias,
                    float* Cf, __bf16* Cb, int Mm, int Nn, int Kk, int relu) {
        dim3 g(Mm / 128, Nn / 128);
        gemm_bf16_kernel<<<g, 256, 0, stream>>>(A, Wt, bias, Cf, Cb, Mm, Nn, Kk, relu);
    };

    // bf16 conversions: activations straight, weights transposed to [N][K]
    cvt((const float*)d_in[0], xb,  MD);
    cvt((const float*)d_in[1], y1b, MD);
    cvt((const float*)d_in[2], y2b, MD);
    for (int m3 = 0; m3 < 3; ++m3) {
        int bse = mha_base[m3];
        cvtT((const float*)d_in[bse + 0], mw[m3].wq, Dm, Dm);
        cvtT((const float*)d_in[bse + 2], mw[m3].wk, Dm, Dm);
        cvtT((const float*)d_in[bse + 4], mw[m3].wv, Dm, Dm);
        cvtT((const float*)d_in[bse + 6], mw[m3].wm, Dm, Dm);
    }
    cvtT(ffn_w1, w1t, Dm, FFd);
    cvtT(ffn_w2, w2t, FFd, Dm);

    auto run_mha = [&](const __bf16* qsrc, const __bf16* kvsrc, const int* maskp,
                       int bse, const MhaW& w, const float* resid,
                       const float* lng, const float* lnb,
                       float* outF, __bf16* outB) {
        gemm(qsrc,  w.wq, (const float*)d_in[bse + 1], nullptr, qb, M, Dm, Dm, 0);
        gemm(kvsrc, w.wk, (const float*)d_in[bse + 3], nullptr, kb, M, Dm, Dm, 0);
        gemm(kvsrc, w.wv, (const float*)d_in[bse + 5], nullptr, vb, M, Dm, Dm, 0);
        attn_kernel<<<dim3(L / 256, 8, Bn), 256, 0, stream>>>(qb, kb, vb, maskp, attnb, L, L);
        gemm(attnb, w.wm, (const float*)d_in[bse + 7], projf, nullptr, M, Dm, Dm, 0);
        ln_kernel<<<M, 256, 0, stream>>>(projf, resid, lng, lnb, outF, outB);
    };

    // x = LN(x + SA(x, x, x_mask))
    run_mha(xb, xb, x_mask, mha_base[0], mw[0], x, ln_g[0], ln_b[0], x1f, x1b);
    // y_x_1 = LN(x + GAA(x, y1, y1_mask))
    run_mha(x1b, y1b, y1_mask, mha_base[1], mw[1], x1f, ln_g[1], ln_b[1], yx1f, nullptr);
    // y_x_2 = LN(x + GAB(x, y2, y2_mask))
    run_mha(x1b, y2b, y2_mask, mha_base[2], mw[2], x1f, ln_g[2], ln_b[2], yx2f, nullptr);

    // x = y_x_1 + y_x_2
    add_kernel<<<(int)((MD + 1023) / 1024), 256, 0, stream>>>(yx1f, yx2f, xuf, xub, MD);

    // FFN: relu(x @ w1 + b1) @ w2 + b2 ; then LN(x + ffn)
    gemm(xub, w1t, ffn_b1, nullptr, h1b, M, FFd, Dm, 1);
    gemm(h1b, w2t, ffn_b2, h2f, nullptr, M, Dm, FFd, 0);
    ln_kernel<<<M, 256, 0, stream>>>(h2f, xuf, ln_g[3], ln_b[3], (float*)d_out, nullptr);
}
